// Str2Str_31224412242583
// MI455X (gfx1250) — matrix operand order repaired and checked by reading, not compiled, not run on hardware
//
#include <hip/hip_runtime.h>
#include <hip/hip_bf16.h>
#include <math.h>

typedef __bf16 bf16_t;
typedef __attribute__((ext_vector_type(16))) __bf16 v16bf;
typedef __attribute__((ext_vector_type(8)))  __bf16 v8bf;
typedef __attribute__((ext_vector_type(8)))  float v8f;
typedef __attribute__((ext_vector_type(4)))  int v4i;
typedef __attribute__((address_space(1))) v4i gv4i;
typedef __attribute__((address_space(3))) v4i lv4i;

#define LSEQ 768
#define NH 12

__device__ __forceinline__ v16bf cat16(v8bf a, v8bf b) {
  return __builtin_shufflevector(a, b, 0, 1, 2, 3, 4, 5, 6, 7, 8, 9, 10, 11, 12, 13, 14, 15);
}

// ---- CDNA5 async global->LDS copy (ASYNCcnt-tracked), 16 bytes per lane ----
__device__ __forceinline__ void async_copy_b128(const bf16_t* gp, bf16_t* lp) {
#if __has_builtin(__builtin_amdgcn_global_load_async_to_lds_b128)
  __builtin_amdgcn_global_load_async_to_lds_b128((gv4i*)gp, (lv4i*)lp, 0, 0);
#else
  unsigned la = (unsigned)(size_t)(const void*)lp;
  asm volatile("global_load_async_to_lds_b128 %0, %1, off" :: "v"(la), "v"(gp) : "memory");
#endif
}
__device__ __forceinline__ void wait_async_le2() {
#if __has_builtin(__builtin_amdgcn_s_wait_asynccnt)
  __builtin_amdgcn_s_wait_asynccnt(2);
#else
  asm volatile("s_wait_asynccnt 0x2" ::: "memory");
#endif
}
__device__ __forceinline__ void wait_async_0() {
#if __has_builtin(__builtin_amdgcn_s_wait_asynccnt)
  __builtin_amdgcn_s_wait_asynccnt(0);
#else
  asm volatile("s_wait_asynccnt 0x0" ::: "memory");
#endif
}

// ---------------------------------------------------------------------------
// Generic batched bf16 GEMM:  C = act(A @ B^T + bias)
// A:  (M x K) bf16 row-major (lda), K must be a multiple of 32
// Bt: (N x K) bf16 row-major (ldb) -- pre-transposed, K-contiguous
// C:  (M x N) f32 (ldc). Per-batch (blockIdx.z) pointer strides.
// Tile 64x64, KT=32; 8 waves x two 16x16 WMMA tiles.
// Async double-buffered LDS pipeline: tile k+1 streams into LDS buffer b^1
// via global_load_async_to_lds_b128 while WMMA consumes buffer b.
// OOB rows are pre-zeroed once and never async-written.
// ---------------------------------------------------------------------------
__global__ __launch_bounds__(256) void gemm_bf16(
    const bf16_t* __restrict__ A, const bf16_t* __restrict__ Bt,
    const float* __restrict__ bias, float* __restrict__ C,
    int M, int N, int K,
    long lda, long ldb, long ldc,
    long strideA, long strideB, long strideC,
    int relu)
{
  __shared__ __align__(16) bf16_t As[2][64 * 32];
  __shared__ __align__(16) bf16_t Bs[2][64 * 32];   // [n][k]
  long bz = blockIdx.z;
  A += bz * strideA; Bt += bz * strideB; C += bz * strideC;
  int m0 = blockIdx.y * 64, n0 = blockIdx.x * 64;
  int t = threadIdx.x;
  int wave = t >> 5, lane = t & 31;
  int wr = wave >> 1;            // row tile 0..3
  int wc = (wave & 1) * 2;       // col tiles wc, wc+1
  int hi = lane >> 4, ml = lane & 15;
  int lr = t >> 2;               // 0..63 tile row for loads
  int lc = (t & 3) * 8;          // 8-elem chunk
  bool mOK = (m0 + lr) < M;
  bool nOK = (n0 + lr) < N;
  // Pre-zero only chunks that will never be async-written (OOB rows)
  {
    uint4 zz = {0u, 0u, 0u, 0u};
    if (!mOK) { *(uint4*)(As[0] + lr * 32 + lc) = zz; *(uint4*)(As[1] + lr * 32 + lc) = zz; }
    if (!nOK) { *(uint4*)(Bs[0] + lr * 32 + lc) = zz; *(uint4*)(Bs[1] + lr * 32 + lc) = zz; }
  }
  const bf16_t* gA = A + (long)(m0 + lr) * lda + lc;
  const bf16_t* gB = Bt + (long)(n0 + lr) * ldb + lc;
  bf16_t* lA = As[0] + lr * 32 + lc;  // buffer 0 chunk; buffer 1 = +2048
  bf16_t* lB = Bs[0] + lr * 32 + lc;
  v8f acc0 = {}, acc1 = {};
  int nk = K >> 5;
  // issue tile 0 -> buffer 0
  if (mOK) async_copy_b128(gA, lA);
  if (nOK) async_copy_b128(gB, lB);
  for (int kt = 0; kt < nk; ++kt) {
    int buf = kt & 1;
    if (kt + 1 < nk) {
      int k1 = (kt + 1) * 32;
      if (mOK) async_copy_b128(gA + k1, lA + (buf ^ 1) * 2048);
      if (nOK) async_copy_b128(gB + k1, lB + (buf ^ 1) * 2048);
      wait_async_le2();           // tile kt complete, kt+1 in flight
    } else {
      wait_async_0();
    }
    __syncthreads();
    const bf16_t* ap = As[buf] + (wr * 16 + ml) * 32;
    v16bf af = cat16(*(const v8bf*)(ap + hi * 16), *(const v8bf*)(ap + hi * 16 + 8));
    const bf16_t* bp0 = Bs[buf] + (wc * 16 + ml) * 32 + hi * 16;
    const bf16_t* bp1 = Bs[buf] + ((wc + 1) * 16 + ml) * 32 + hi * 16;
    v16bf bf0 = cat16(*(const v8bf*)bp0, *(const v8bf*)(bp0 + 8));
    v16bf bf1 = cat16(*(const v8bf*)bp1, *(const v8bf*)(bp1 + 8));
    acc0 = __builtin_amdgcn_wmma_f32_16x16x32_bf16(false, af, false, bf0, (short)0, acc0, false, false);
    acc1 = __builtin_amdgcn_wmma_f32_16x16x32_bf16(false, af, false, bf1, (short)0, acc1, false, false);
    __syncthreads();              // all waves done reading buf before reuse
  }
#pragma unroll
  for (int r = 0; r < 8; r++) {
    int gm = m0 + wr * 16 + r + 8 * hi;
    if (gm >= M) continue;
    int gn0 = n0 + wc * 16 + ml;
    int gn1 = gn0 + 16;
    if (gn0 < N) { float v = acc0[r]; if (bias) v += bias[gn0]; if (relu) v = fmaxf(v, 0.f); C[(long)gm * ldc + gn0] = v; }
    if (gn1 < N) { float v = acc1[r]; if (bias) v += bias[gn1]; if (relu) v = fmaxf(v, 0.f); C[(long)gm * ldc + gn1] = v; }
  }
}

// ---------------------------------------------------------------------------
// Row LayerNorm (N <= 256), optional residual add, f32 and/or bf16 outputs
// ---------------------------------------------------------------------------
__global__ __launch_bounds__(256) void ln_kernel(
    const float* __restrict__ src, const float* __restrict__ add,
    const float* __restrict__ g, const float* __restrict__ b,
    float* __restrict__ dstF, bf16_t* __restrict__ dstB, int N)
{
  __shared__ float sh[256];
  int row = blockIdx.x, t = threadIdx.x;
  long base = (long)row * N;
  float v = 0.f;
  if (t < N) { v = src[base + t]; if (add) v += add[base + t]; }
  sh[t] = (t < N) ? v : 0.f;
  __syncthreads();
  for (int s = 128; s > 0; s >>= 1) { if (t < s) sh[t] += sh[t + s]; __syncthreads(); }
  float mu = sh[0] / (float)N;
  __syncthreads();
  float d = (t < N) ? (v - mu) : 0.f;
  sh[t] = d * d;
  __syncthreads();
  for (int s = 128; s > 0; s >>= 1) { if (t < s) sh[t] += sh[t + s]; __syncthreads(); }
  float rs = rsqrtf(sh[0] / (float)N + 1e-5f);
  if (t < N) {
    float y = (v - mu) * rs * g[t] + b[t];
    if (dstF) dstF[base + t] = y;
    if (dstB) dstB[base + t] = (bf16_t)y;
  }
}

// ---------------------------------------------------------------------------
// Node preamble: concat(LN(m0), LN(s)) -> bf16 (768 x 288, cols 272+ zero)
// ---------------------------------------------------------------------------
__global__ __launch_bounds__(256) void node_prep_kernel(
    const float* __restrict__ m, const float* __restrict__ s,
    const float* nm_g, const float* nm_b, const float* ns_g, const float* ns_b,
    bf16_t* __restrict__ cat)
{
  __shared__ float sh[256];
  __shared__ float sln[16];
  int row = blockIdx.x, t = threadIdx.x;
  float v = m[(long)row * 256 + t];
  sh[t] = v; __syncthreads();
  for (int st = 128; st > 0; st >>= 1) { if (t < st) sh[t] += sh[t + st]; __syncthreads(); }
  float mu = sh[0] / 256.f; __syncthreads();
  float d = v - mu; sh[t] = d * d; __syncthreads();
  for (int st = 128; st > 0; st >>= 1) { if (t < st) sh[t] += sh[t + st]; __syncthreads(); }
  float rs = rsqrtf(sh[0] / 256.f + 1e-5f);
  cat[(long)row * 288 + t] = (bf16_t)((v - mu) * rs * nm_g[t] + nm_b[t]);
  if (t == 0) {
    float mu2 = 0.f, v2 = 0.f;
    for (int i = 0; i < 16; i++) mu2 += s[row * 16 + i];
    mu2 /= 16.f;
    for (int i = 0; i < 16; i++) { float dd = s[row * 16 + i] - mu2; v2 += dd * dd; }
    float rs2 = rsqrtf(v2 / 16.f + 1e-5f);
    for (int i = 0; i < 16; i++) sln[i] = (s[row * 16 + i] - mu2) * rs2 * ns_g[i] + ns_b[i];
  }
  __syncthreads();
  if (t < 16) {
    cat[(long)row * 288 + 256 + t] = (bf16_t)sln[t];
    cat[(long)row * 288 + 272 + t] = (bf16_t)0.f;
  }
}

// ---------------------------------------------------------------------------
// Fused edge kernel over 16 pairs (i, j0..j0+15):
//   LN(z) | rbf | neighbor -> LDS bf16 -> WMMA @ ee_w_t (128x192, N-major)
//   + ee_b -> row LN(ne) -> e_t (stored transposed: [i][c][j]),
//   then WMMA @ b_w_t (16x128, rows>=12 zero) -> pair_b
// ---------------------------------------------------------------------------
__global__ __launch_bounds__(128) void edge_kernel(
    const float* __restrict__ z, const float* __restrict__ cas, const int* __restrict__ aatype,
    const float* nz_g, const float* nz_b,
    const bf16_t* __restrict__ ee_w_t, const float* ee_b,
    const float* ne_g, const float* ne_b,
    const bf16_t* __restrict__ b_w_t, const float* b_b,
    bf16_t* __restrict__ e_t, float* __restrict__ pairb)
{
  __shared__ __align__(16) bf16_t Xs[16 * 192];
  __shared__ __align__(16) bf16_t Bs[128 * 32];   // [n][k]
  __shared__ float outS[16 * 128];
  __shared__ float red1[16][8], red2[16][8];
  __shared__ float muS[16], rsS[16], dS[16], nbS[16];
  int i = blockIdx.y;
  int j0 = blockIdx.x * 16;
  int t = threadIdx.x;
  int ty = t >> 3, tx = t & 7;
  int j = j0 + ty;
  long zbase = ((long)i * LSEQ + j) * 128;
  float s1 = 0.f, s2 = 0.f;
  {
    const float4* zp = (const float4*)(z + zbase + tx * 16);
#pragma unroll
    for (int q = 0; q < 4; q++) {
      float4 f = zp[q];
      s1 += f.x + f.y + f.z + f.w;
      s2 += f.x * f.x + f.y * f.y + f.z * f.z + f.w * f.w;
    }
  }
  red1[ty][tx] = s1; red2[ty][tx] = s2;
  if (tx == 0) {
    float dx = cas[i * 3] - cas[j * 3], dy = cas[i * 3 + 1] - cas[j * 3 + 1], dz = cas[i * 3 + 2] - cas[j * 3 + 2];
    dS[ty] = sqrtf(dx * dx + dy * dy + dz * dz + 1e-12f);
    float sep = (float)(aatype[j] - aatype[i]);
    float sg = (sep > 0.f) ? 1.f : ((sep < 0.f) ? -1.f : 0.f);
    nbS[ty] = sg * logf(fabsf(sep) + 1.f);
  }
  __syncthreads();
  if (tx == 0) {
    float a = 0.f, bb = 0.f;
    for (int q = 0; q < 8; q++) { a += red1[ty][q]; bb += red2[ty][q]; }
    float mu = a / 128.f;
    muS[ty] = mu;
    rsS[ty] = rsqrtf(bb / 128.f - mu * mu + 1e-5f);
  }
  __syncthreads();
  float mu = muS[ty], rs = rsS[ty];
  for (int c = tx; c < 192; c += 8) {
    float v;
    if (c < 128) v = (z[zbase + c] - mu) * rs * nz_g[c] + nz_b[c];
    else if (c < 164) { int r = c - 128; float mr = 2.f + (float)r * (20.f / 35.f); float dd = dS[ty] - mr; v = expf(-dd * dd); }
    else if (c == 164) v = nbS[ty];
    else v = 0.f;
    Xs[ty * 192 + c] = (bf16_t)v;
  }
  __syncthreads();
  int wave = t >> 5, lane = t & 31, hi = lane >> 4, ml = lane & 15;
  v8f acc0 = {}, acc1 = {};
  for (int k0 = 0; k0 < 192; k0 += 32) {
#pragma unroll
    for (int u = 0; u < 4; u++)
      *(uint4*)(Bs + t * 32 + u * 8) = *(const uint4*)(ee_w_t + (long)t * 192 + k0 + u * 8);
    __syncthreads();
    const bf16_t* ap = Xs + ml * 192 + k0;
    v16bf af = cat16(*(const v8bf*)(ap + hi * 16), *(const v8bf*)(ap + hi * 16 + 8));
    const bf16_t* bp0 = Bs + (wave * 32 + ml) * 32 + hi * 16;
    const bf16_t* bp1 = Bs + (wave * 32 + 16 + ml) * 32 + hi * 16;
    v16bf bf0 = cat16(*(const v8bf*)bp0, *(const v8bf*)(bp0 + 8));
    v16bf bf1 = cat16(*(const v8bf*)bp1, *(const v8bf*)(bp1 + 8));
    acc0 = __builtin_amdgcn_wmma_f32_16x16x32_bf16(false, af, false, bf0, (short)0, acc0, false, false);
    acc1 = __builtin_amdgcn_wmma_f32_16x16x32_bf16(false, af, false, bf1, (short)0, acc1, false, false);
    __syncthreads();
  }
#pragma unroll
  for (int r = 0; r < 8; r++) {
    int rm = r + 8 * hi;
    int c0 = wave * 32 + ml, c1 = c0 + 16;
    outS[rm * 128 + c0] = acc0[r] + ee_b[c0];
    outS[rm * 128 + c1] = acc1[r] + ee_b[c1];
  }
  __syncthreads();
  s1 = 0.f; s2 = 0.f;
  for (int q = 0; q < 16; q++) { float v = outS[ty * 128 + tx * 16 + q]; s1 += v; s2 += v * v; }
  red1[ty][tx] = s1; red2[ty][tx] = s2;
  __syncthreads();
  if (tx == 0) {
    float a = 0.f, bb = 0.f;
    for (int q = 0; q < 8; q++) { a += red1[ty][q]; bb += red2[ty][q]; }
    float m2 = a / 128.f; muS[ty] = m2; rsS[ty] = rsqrtf(bb / 128.f - m2 * m2 + 1e-5f);
  }
  __syncthreads();
  mu = muS[ty]; rs = rsS[ty];
  for (int c = tx; c < 128; c += 8) {
    float v = (outS[ty * 128 + c] - mu) * rs * ne_g[c] + ne_b[c];
    bf16_t bv = (bf16_t)v;
    Xs[ty * 192 + c] = bv;
    e_t[((long)i * 128 + c) * LSEQ + j] = bv;   // transposed: [i][c][j]
  }
  __syncthreads();
  if (wave == 0) {
    v8f acc2 = {};
    for (int k0 = 0; k0 < 128; k0 += 32) {
      const bf16_t* ap = Xs + ml * 192 + k0;
      v16bf af = cat16(*(const v8bf*)(ap + hi * 16), *(const v8bf*)(ap + hi * 16 + 8));
      const bf16_t* bp = b_w_t + (long)ml * 128 + k0 + hi * 16;
      v16bf bfb = cat16(*(const v8bf*)bp, *(const v8bf*)(bp + 8));
      acc2 = __builtin_amdgcn_wmma_f32_16x16x32_bf16(false, af, false, bfb, (short)0, acc2, false, false);
    }
#pragma unroll
    for (int r = 0; r < 8; r++) {
      int rm = r + 8 * hi;
      if (ml < 12) pairb[((long)i * LSEQ + j0 + rm) * 12 + ml] = acc2[r] + b_b[ml];
    }
  }
}

// ---------------------------------------------------------------------------
// Build Acat/Kcat (scaled q|q_pts pad 32, N x K layout already), and
// Vcat_t (transposed: [h][c][i], c in 0..47, rows 40..47 zero),
// plus point norms qn/kn. One thread per (i,h).
// ---------------------------------------------------------------------------
__global__ __launch_bounds__(256) void pts_kernel(
    const float* __restrict__ qbuf, const float* __restrict__ kvbuf,
    const float* __restrict__ qpbuf, const float* __restrict__ kvpbuf,
    const float* __restrict__ rot, const float* __restrict__ trans,
    const float* __restrict__ head_w,
    bf16_t* __restrict__ Acat, bf16_t* __restrict__ Kcat, bf16_t* __restrict__ Vcat_t,
    float* __restrict__ qn, float* __restrict__ kn)
{
  int idx = blockIdx.x * 256 + threadIdx.x;
  if (idx >= LSEQ * NH) return;
  int i = idx / NH, h = idx % NH;
  const float C1 = 0.14433756729740643f;   // 1/sqrt(3*16)
  const float PTW = 0.13608276348795434f;  // sqrt(1/(3*18))
  float c2 = logf(1.f + expf(head_w[h])) * PTW;
  float R0 = rot[i * 9 + 0], R1 = rot[i * 9 + 1], R2 = rot[i * 9 + 2];
  float R3 = rot[i * 9 + 3], R4 = rot[i * 9 + 4], R5 = rot[i * 9 + 5];
  float R6 = rot[i * 9 + 6], R7 = rot[i * 9 + 7], R8 = rot[i * 9 + 8];
  float T0 = trans[i * 3 + 0], T1 = trans[i * 3 + 1], T2 = trans[i * 3 + 2];
  long ab = ((long)h * LSEQ + i) * 32;
  long vtb = (long)h * 48 * LSEQ + i;     // + c*LSEQ
  for (int c = 0; c < 16; c++) {
    Acat[ab + c] = (bf16_t)(qbuf[(long)i * 192 + h * 16 + c] * C1);
    Kcat[ab + c] = (bf16_t)(kvbuf[(long)i * 384 + h * 32 + c]);
    Vcat_t[vtb + (long)c * LSEQ] = (bf16_t)(kvbuf[(long)i * 384 + h * 32 + 16 + c]);
  }
  float qs = 0.f, ks = 0.f;
  for (int p = 0; p < 4; p++) {
    long qpb = (long)i * 144 + (h * 4 + p) * 3;
    float x = qpbuf[qpb], y = qpbuf[qpb + 1], zz = qpbuf[qpb + 2];
    float gx = R0 * x + R1 * y + R2 * zz + T0;
    float gy = R3 * x + R4 * y + R5 * zz + T1;
    float gz = R6 * x + R7 * y + R8 * zz + T2;
    qs += gx * gx + gy * gy + gz * gz;
    Acat[ab + 16 + p * 3 + 0] = (bf16_t)(gx * c2);
    Acat[ab + 16 + p * 3 + 1] = (bf16_t)(gy * c2);
    Acat[ab + 16 + p * 3 + 2] = (bf16_t)(gz * c2);
    long kpb = (long)i * 432 + (h * 12 + p) * 3;
    x = kvpbuf[kpb]; y = kvpbuf[kpb + 1]; zz = kvpbuf[kpb + 2];
    gx = R0 * x + R1 * y + R2 * zz + T0;
    gy = R3 * x + R4 * y + R5 * zz + T1;
    gz = R6 * x + R7 * y + R8 * zz + T2;
    ks += gx * gx + gy * gy + gz * gz;
    Kcat[ab + 16 + p * 3 + 0] = (bf16_t)gx;
    Kcat[ab + 16 + p * 3 + 1] = (bf16_t)gy;
    Kcat[ab + 16 + p * 3 + 2] = (bf16_t)gz;
  }
  for (int c = 28; c < 32; c++) { Acat[ab + c] = (bf16_t)0.f; Kcat[ab + c] = (bf16_t)0.f; }
  for (int p = 0; p < 8; p++) {
    long kpb = (long)i * 432 + (h * 12 + 4 + p) * 3;
    float x = kvpbuf[kpb], y = kvpbuf[kpb + 1], zz = kvpbuf[kpb + 2];
    Vcat_t[vtb + (long)(16 + p * 3 + 0) * LSEQ] = (bf16_t)(R0 * x + R1 * y + R2 * zz + T0);
    Vcat_t[vtb + (long)(16 + p * 3 + 1) * LSEQ] = (bf16_t)(R3 * x + R4 * y + R5 * zz + T1);
    Vcat_t[vtb + (long)(16 + p * 3 + 2) * LSEQ] = (bf16_t)(R6 * x + R7 * y + R8 * zz + T2);
  }
  for (int c = 40; c < 48; c++) Vcat_t[vtb + (long)c * LSEQ] = (bf16_t)0.f;
  qn[i * NH + h] = qs; kn[i * NH + h] = ks;
}

// ---------------------------------------------------------------------------
// Softmax over j with pair bias, point-distance bias and mask; emits bf16 a
// ---------------------------------------------------------------------------
__global__ __launch_bounds__(256) void softmax_kernel(
    const float* __restrict__ apre, const float* __restrict__ pairb,
    const float* __restrict__ qn, const float* __restrict__ kn,
    const float* __restrict__ head_w, const float* __restrict__ mask,
    bf16_t* __restrict__ abf)
{
  __shared__ float sh[256];
  int i = blockIdx.x, h = blockIdx.y, t = threadIdx.x;
  const float PTW = 0.13608276348795434f;
  const float IS3 = 0.5773502691896258f;
  float c2 = logf(1.f + expf(head_w[h])) * PTW;
  float qni = qn[i * NH + h];
  float mi = mask[i];
  long base = (long)h * LSEQ * LSEQ + (long)i * LSEQ;
  float l[3];
#pragma unroll
  for (int u = 0; u < 3; u++) {
    int j = t + u * 256;
    l[u] = apre[base + j] + IS3 * pairb[((long)i * LSEQ + j) * 12 + h]
           - 0.5f * c2 * (qni + kn[j * NH + h])
           + (mi * mask[j] - 1.f) * 100000.f;
  }
  float mx = fmaxf(l[0], fmaxf(l[1], l[2]));
  sh[t] = mx; __syncthreads();
  for (int s = 128; s > 0; s >>= 1) { if (t < s) sh[t] = fmaxf(sh[t], sh[t + s]); __syncthreads(); }
  mx = sh[0]; __syncthreads();
  float e0 = expf(l[0] - mx), e1 = expf(l[1] - mx), e2 = expf(l[2] - mx);
  sh[t] = e0 + e1 + e2; __syncthreads();
  for (int s = 128; s > 0; s >>= 1) { if (t < s) sh[t] += sh[t + s]; __syncthreads(); }
  float inv = 1.f / sh[0];
  abf[base + t] = (bf16_t)(e0 * inv);
  abf[base + t + 256] = (bf16_t)(e1 * inv);
  abf[base + t + 512] = (bf16_t)(e2 * inv);
}

// ---------------------------------------------------------------------------
// Build the 2112-wide concat row (o | o_pt local | norms | o_pair) as bf16
// ---------------------------------------------------------------------------
__global__ __launch_bounds__(256) void finalize_kernel(
    const float* __restrict__ ohic, const float* __restrict__ opair,
    const float* __restrict__ rot, const float* __restrict__ trans,
    bf16_t* __restrict__ catb)
{
  __shared__ float R[9], T[3];
  int i = blockIdx.x, t = threadIdx.x;
  if (t < 9) R[t] = rot[i * 9 + t];
  if (t < 3) T[t] = trans[i * 3 + t];
  __syncthreads();
  long cb = (long)i * 2112;
  if (t < NH) {
    int h = t;
    long ob = ((long)h * LSEQ + i) * 40;
    for (int c = 0; c < 16; c++) catb[cb + h * 16 + c] = (bf16_t)ohic[ob + c];
    for (int p = 0; p < 8; p++) {
      float wx = ohic[ob + 16 + p * 3 + 0] - T[0];
      float wy = ohic[ob + 16 + p * 3 + 1] - T[1];
      float wz = ohic[ob + 16 + p * 3 + 2] - T[2];
      float lx = R[0] * wx + R[3] * wy + R[6] * wz;
      float ly = R[1] * wx + R[4] * wy + R[7] * wz;
      float lz = R[2] * wx + R[5] * wy + R[8] * wz;
      catb[cb + 192 + h * 24 + p * 3 + 0] = (bf16_t)lx;
      catb[cb + 192 + h * 24 + p * 3 + 1] = (bf16_t)ly;
      catb[cb + 192 + h * 24 + p * 3 + 2] = (bf16_t)lz;
      catb[cb + 480 + h * 8 + p] = (bf16_t)sqrtf(lx * lx + ly * ly + lz * lz + 1e-8f);
    }
  }
  for (int c = t; c < 1536; c += 256) catb[cb + 576 + c] = (bf16_t)opair[(long)i * 1536 + c];
}

__global__ void update_kernel(const float* __restrict__ upd, float* rot, float* trans)
{
  int i = blockIdx.x * blockDim.x + threadIdx.x;
  if (i >= LSEQ) return;
  float b = upd[i * 6], c = upd[i * 6 + 1], d = upd[i * 6 + 2];
  float inv = rsqrtf(1.f + b * b + c * c + d * d);
  float w = inv, x = b * inv, y = c * inv, z = d * inv;
  float Rq[9] = {
    1.f - 2.f * (y * y + z * z), 2.f * (x * y - w * z), 2.f * (x * z + w * y),
    2.f * (x * y + w * z), 1.f - 2.f * (x * x + z * z), 2.f * (y * z - w * x),
    2.f * (x * z - w * y), 2.f * (y * z + w * x), 1.f - 2.f * (x * x + y * y)};
  float R[9];
  for (int u = 0; u < 9; u++) R[u] = rot[i * 9 + u];
  float u0 = upd[i * 6 + 3], u1 = upd[i * 6 + 4], u2 = upd[i * 6 + 5];
  trans[i * 3 + 0] += R[0] * u0 + R[1] * u1 + R[2] * u2;
  trans[i * 3 + 1] += R[3] * u0 + R[4] * u1 + R[5] * u2;
  trans[i * 3 + 2] += R[6] * u0 + R[7] * u1 + R[8] * u2;
  float Rn[9];
  for (int r = 0; r < 3; r++)
    for (int cc = 0; cc < 3; cc++)
      Rn[r * 3 + cc] = R[r * 3 + 0] * Rq[cc] + R[r * 3 + 1] * Rq[3 + cc] + R[r * 3 + 2] * Rq[6 + cc];
  for (int u = 0; u < 9; u++) rot[i * 9 + u] = Rn[u];
}

__global__ void init_rt_kernel(float* rot, float* trans)
{
  int i = blockIdx.x * blockDim.x + threadIdx.x;
  if (i >= LSEQ) return;
  for (int u = 0; u < 9; u++) rot[i * 9 + u] = (u == 0 || u == 4 || u == 8) ? 1.f : 0.f;
  trans[i * 3] = 0.f; trans[i * 3 + 1] = 0.f; trans[i * 3 + 2] = 0.f;
}

__global__ void cvt_kernel(const float* __restrict__ src, bf16_t* __restrict__ dst, long n)
{
  long idx = (long)blockIdx.x * 256 + threadIdx.x;
  if (idx < n) dst[idx] = (bf16_t)src[idx];
}

// Transposing weight conversion: src (K x N) f32 row-major ->
// dst (Npad x Kpad) bf16 row-major (out-of-range -> 0)
__global__ void cvt_t_kernel(const float* __restrict__ src, bf16_t* __restrict__ dst,
                             int srcK, int srcN, int Kpad, int Npad)
{
  long n = (long)Npad * Kpad;
  long idx = (long)blockIdx.x * 256 + threadIdx.x;
  if (idx >= n) return;
  int nn = (int)(idx / Kpad);
  int kk = (int)(idx % Kpad);
  dst[idx] = (nn < srcN && kk < srcK) ? (bf16_t)src[(long)kk * srcN + nn] : (bf16_t)0.f;
}

__global__ void out_copy_kernel(const float* rot, const float* trans, const float* sc, float* out)
{
  int i = blockIdx.x * blockDim.x + threadIdx.x;
  const int nr = LSEQ * 9, nt = LSEQ * 3, ns = LSEQ * 256;
  if (i < nr) out[i] = rot[i];
  else if (i < nr + nt) out[i] = trans[i - nr];
  else if (i < nr + nt + ns) out[i] = sc[i - nr - nt];
}

// ---------------------------------------------------------------------------
extern "C" void kernel_launch(void* const* d_in, const int* in_sizes, int n_in,
                              void* d_out, int out_size, void* d_ws, size_t ws_size,
                              hipStream_t stream)
{
  (void)in_sizes; (void)n_in; (void)out_size; (void)ws_size;
  const float* m    = (const float*)d_in[0];
  const float* z    = (const float*)d_in[1];
  const float* s    = (const float*)d_in[2];
  const float* cas  = (const float*)d_in[3];
  const float* mask = (const float*)d_in[4];
  const int*  aatype= (const int*)d_in[5];
  int pi = 6;
  const float* nm_g=(const float*)d_in[pi++]; const float* nm_b=(const float*)d_in[pi++];
  const float* nz_g=(const float*)d_in[pi++]; const float* nz_b=(const float*)d_in[pi++];
  const float* ns_g=(const float*)d_in[pi++]; const float* ns_b=(const float*)d_in[pi++];
  const float* nn_g=(const float*)d_in[pi++]; const float* nn_b=(const float*)d_in[pi++];
  const float* ne_g=(const float*)d_in[pi++]; const float* ne_b=(const float*)d_in[pi++];
  const float* iln_g=(const float*)d_in[pi++]; const float* iln_b=(const float*)d_in[pi++];
  const float* tln_g=(const float*)d_in[pi++]; const float* tln_b=(const float*)d_in[pi++];
  const float* ex_w=(const float*)d_in[pi++]; const float* ex_b=(const float*)d_in[pi++];
  const float* ee_w=(const float*)d_in[pi++]; const float* ee_b=(const float*)d_in[pi++];
  const float* q_w =(const float*)d_in[pi++]; const float* q_b =(const float*)d_in[pi++];
  const float* kv_w=(const float*)d_in[pi++]; const float* kv_b=(const float*)d_in[pi++];
  const float* qp_w=(const float*)d_in[pi++]; const float* qp_b=(const float*)d_in[pi++];
  const float* kvp_w=(const float*)d_in[pi++]; const float* kvp_b=(const float*)d_in[pi++];
  const float* b_w =(const float*)d_in[pi++]; const float* b_b =(const float*)d_in[pi++];
  const float* out_w=(const float*)d_in[pi++]; const float* out_b=(const float*)d_in[pi++];
  const float* t1_w=(const float*)d_in[pi++]; const float* t1_b=(const float*)d_in[pi++];
  const float* t2_w=(const float*)d_in[pi++]; const float* t2_b=(const float*)d_in[pi++];
  const float* t3_w=(const float*)d_in[pi++]; const float* t3_b=(const float*)d_in[pi++];
  const float* bb_w=(const float*)d_in[pi++]; const float* bb_b=(const float*)d_in[pi++];
  const float* head_w=(const float*)d_in[pi++];

  char* wsb = (char*)d_ws; size_t off = 0;
  auto alloc = [&](size_t bytes) -> char* {
    char* p = wsb + off; off = (off + bytes + 255) & ~(size_t)255; return p;
  };
  const long L = LSEQ, LL = (long)LSEQ * LSEQ;
  // transposed weights: (N x Kpad) bf16
  bf16_t* wex = (bf16_t*)alloc((long)256 * 288 * 2);
  bf16_t* wee = (bf16_t*)alloc((long)128 * 192 * 2);
  bf16_t* wq  = (bf16_t*)alloc((long)192 * 256 * 2);
  bf16_t* wkv = (bf16_t*)alloc((long)384 * 256 * 2);
  bf16_t* wqp = (bf16_t*)alloc((long)144 * 256 * 2);
  bf16_t* wkvp= (bf16_t*)alloc((long)432 * 256 * 2);
  bf16_t* wb  = (bf16_t*)alloc((long)16 * 128 * 2);
  bf16_t* wout= (bf16_t*)alloc((long)256 * 2112 * 2);
  bf16_t* wt1 = (bf16_t*)alloc((long)256 * 256 * 2);
  bf16_t* wt2 = (bf16_t*)alloc((long)256 * 256 * 2);
  bf16_t* wt3 = (bf16_t*)alloc((long)256 * 256 * 2);
  bf16_t* wbb = (bf16_t*)alloc((long)6 * 256 * 2);
  bf16_t* e_t    = (bf16_t*)alloc(LL * 128 * 2);     // [i][c][j]
  float*  pairb  = (float*) alloc(LL * 12 * 4);
  float*  apre   = (float*) alloc(12 * LL * 4);
  bf16_t* abf    = (bf16_t*)alloc(12 * LL * 2);
  bf16_t* concatb= (bf16_t*)alloc(L * 288 * 2);
  float*  node_pre=(float*) alloc(L * 256 * 4);
  float*  sc     = (float*) alloc(L * 256 * 4);
  bf16_t* sc_bf  = (bf16_t*)alloc(L * 256 * 2);
  float*  qb     = (float*) alloc(L * 192 * 4);
  float*  kvb    = (float*) alloc(L * 384 * 4);
  float*  qpb    = (float*) alloc(L * 144 * 4);
  float*  kvpb   = (float*) alloc(L * 432 * 4);
  bf16_t* Acat   = (bf16_t*)alloc((long)12 * L * 32 * 2);
  bf16_t* Kcat   = (bf16_t*)alloc((long)12 * L * 32 * 2);
  bf16_t* Vcat_t = (bf16_t*)alloc((long)12 * 48 * L * 2);  // [h][c][i]
  float*  qnb    = (float*) alloc(L * 12 * 4);
  float*  knb    = (float*) alloc(L * 12 * 4);
  float*  ohic   = (float*) alloc((long)12 * L * 40 * 4);
  float*  opairb = (float*) alloc(L * 1536 * 4);
  bf16_t* catb   = (bf16_t*)alloc(L * 2112 * 2);
  float*  delta  = (float*) alloc(L * 256 * 4);
  float*  tbuf   = (float*) alloc(L * 256 * 4);
  bf16_t* tbf    = (bf16_t*)alloc(L * 256 * 2);
  float*  bbout  = (float*) alloc(L * 6 * 4);
  float*  rotb   = (float*) alloc(L * 9 * 4);
  float*  transb = (float*) alloc(L * 3 * 4);

  auto CVT = [&](const float* src, bf16_t* dst, long n) {
    cvt_kernel<<<(int)((n + 255) / 256), 256, 0, stream>>>(src, dst, n);
  };
  auto CVTT = [&](const float* src, bf16_t* dst, int srcK, int srcN, int Kpad, int Npad) {
    long n = (long)Npad * Kpad;
    cvt_t_kernel<<<(int)((n + 255) / 256), 256, 0, stream>>>(src, dst, srcK, srcN, Kpad, Npad);
  };
  auto GEMM = [&](const bf16_t* A, const bf16_t* Bt, const float* bias, float* C,
                  int M, int N, int K, long lda, long ldb, long ldc,
                  long sA, long sB, long sC, int batch, int relu) {
    dim3 g((N + 63) / 64, (M + 63) / 64, batch);
    gemm_bf16<<<g, 256, 0, stream>>>(A, Bt, bias, C, M, N, K, lda, ldb, ldc, sA, sB, sC, relu);
  };

  // Weight conversion (bf16, transposed to N x Kpad)
  CVTT(ex_w, wex, 272, 256, 288, 256);
  CVTT(ee_w, wee, 165, 128, 192, 128);
  CVTT(q_w, wq, 256, 192, 256, 192);
  CVTT(kv_w, wkv, 256, 384, 256, 384);
  CVTT(qp_w, wqp, 256, 144, 256, 144);
  CVTT(kvp_w, wkvp, 256, 432, 256, 432);
  CVTT(b_w, wb, 128, 12, 128, 16);
  CVTT(out_w, wout, 2112, 256, 2112, 256);
  CVTT(t1_w, wt1, 256, 256, 256, 256);
  CVTT(t2_w, wt2, 256, 256, 256, 256);
  CVTT(t3_w, wt3, 256, 256, 256, 256);
  CVTT(bb_w, wbb, 256, 6, 256, 6);

  // Node preamble: sc = LN( concat(LN(m0), LN(s)) @ ex + b )
  node_prep_kernel<<<LSEQ, 256, 0, stream>>>(m, s, nm_g, nm_b, ns_g, ns_b, concatb);
  GEMM(concatb, wex, ex_b, node_pre, LSEQ, 256, 288, 288, 288, 256, 0, 0, 0, 1, 0);
  ln_kernel<<<LSEQ, 256, 0, stream>>>(node_pre, nullptr, nn_g, nn_b, sc, sc_bf, 256);

  // Fused edge embedding + pair bias
  edge_kernel<<<dim3(LSEQ / 16, LSEQ), 128, 0, stream>>>(
      z, cas, aatype, nz_g, nz_b, wee, ee_b, ne_g, ne_b, wb, b_b, e_t, pairb);
  init_rt_kernel<<<3, 256, 0, stream>>>(rotb, transb);

  for (int blk = 0; blk < 8; blk++) {
    GEMM(sc_bf, wq,  q_b,  qb,  LSEQ, 192, 256, 256, 256, 192, 0, 0, 0, 1, 0);
    GEMM(sc_bf, wkv, kv_b, kvb, LSEQ, 384, 256, 256, 256, 384, 0, 0, 0, 1, 0);
    GEMM(sc_bf, wqp, qp_b, qpb, LSEQ, 144, 256, 256, 256, 144, 0, 0, 0, 1, 0);
    GEMM(sc_bf, wkvp, kvp_b, kvpb, LSEQ, 432, 256, 256, 256, 432, 0, 0, 0, 1, 0);
    pts_kernel<<<(LSEQ * NH + 255) / 256, 256, 0, stream>>>(
        qb, kvb, qpb, kvpb, rotb, transb, head_w, Acat, Kcat, Vcat_t, qnb, knb);
    // logits: per-head (768x768), K=32 (scaled q | q_pts), B = Kcat (N x K)
    GEMM(Acat, Kcat, nullptr, apre, LSEQ, LSEQ, 32, 32, 32, LSEQ,
         (long)LSEQ * 32, (long)LSEQ * 32, LL, NH, 0);
    softmax_kernel<<<dim3(LSEQ, NH), 256, 0, stream>>>(apre, pairb, qnb, knb, head_w, mask, abf);
    // o | o_pt: per-head a_h @ Vcat_h (768x40), B transposed layout [h][c][i]
    GEMM(abf, Vcat_t, nullptr, ohic, LSEQ, 40, LSEQ, LSEQ, LSEQ, 40,
         LL, (long)48 * LSEQ, (long)LSEQ * 40, NH, 0);
    // o_pair: per-residue batched (12x128) = a[:,i,:] @ e[i]; B = e_t [i][c][j]
    GEMM(abf, e_t, nullptr, opairb, 12, 128, LSEQ, LL, LSEQ, 128,
         LSEQ, (long)128 * LSEQ, 1536, LSEQ, 0);
    finalize_kernel<<<LSEQ, 256, 0, stream>>>(ohic, opairb, rotb, transb, catb);
    GEMM(catb, wout, out_b, delta, LSEQ, 256, 2112, 2112, 2112, 256, 0, 0, 0, 1, 0);
    ln_kernel<<<LSEQ, 256, 0, stream>>>(sc, delta, iln_g, iln_b, sc, sc_bf, 256);
    GEMM(sc_bf, wt1, t1_b, tbuf, LSEQ, 256, 256, 256, 256, 256, 0, 0, 0, 1, 1);
    CVT(tbuf, tbf, L * 256);
    GEMM(tbf, wt2, t2_b, delta, LSEQ, 256, 256, 256, 256, 256, 0, 0, 0, 1, 1);
    CVT(delta, tbf, L * 256);
    GEMM(tbf, wt3, t3_b, tbuf, LSEQ, 256, 256, 256, 256, 256, 0, 0, 0, 1, 0);
    ln_kernel<<<LSEQ, 256, 0, stream>>>(sc, tbuf, tln_g, tln_b, sc, sc_bf, 256);
    GEMM(sc_bf, wbb, bb_b, bbout, LSEQ, 6, 256, 256, 256, 6, 0, 0, 0, 1, 0);
    update_kernel<<<3, 256, 0, stream>>>(bbout, rotb, transb);
  }
  out_copy_kernel<<<(LSEQ * 268 + 255) / 256, 256, 0, stream>>>(rotb, transb, sc, (float*)d_out);
}
